// AttentionBlock_90744069029951
// MI455X (gfx1250) — matrix operand order repaired and checked by reading, not compiled, not run on hardware
//
#include <hip/hip_runtime.h>

#define B_  2
#define S_  2048
#define D_  2048
#define H_  16
#define HD_ 128

typedef __attribute__((ext_vector_type(16))) _Float16 v16h;
typedef __attribute__((ext_vector_type(8)))  _Float16 v8h;
typedef __attribute__((ext_vector_type(4)))  _Float16 v4h;
typedef __attribute__((ext_vector_type(8)))  float    v8f;
typedef __attribute__((ext_vector_type(4)))  float    v4f;

__device__ __forceinline__ v16h combine16(v8h lo, v8h hi) {
  return __builtin_shufflevector(lo, hi, 0,1,2,3,4,5,6,7,8,9,10,11,12,13,14,15);
}

__device__ __forceinline__ v8f wmma_f16(v16h a, v16h b, v8f c) {
  // D = A(16x32 f16) * B(32x16 f16) + C(16x16 f32)
  return __builtin_amdgcn_wmma_f32_16x16x32_f16(false, a, false, b, (short)0, c, false, false);
}

// ---------------------------------------------------------------------------
// WMMA GEMM: C[M,N] = A[M,K] * Bw[K,N], LDS double-buffered.
//  - A: row-major, float or _Float16;  Bw: row-major float;  C: _Float16 or float
//  - Block tile 128x128, 8 waves (4 along M x 2 along N), wave tile 32x64
//  - K-step 32; fp32 -> f16 convert at fetch; B stored transposed [n][k]
//  - M%128==0, N%128==0, K%32==0 (true for all uses here)
// ---------------------------------------------------------------------------
template<typename AT, typename CT>
__global__ __launch_bounds__(256) void gemm_wmma_kernel(
    const AT* __restrict__ A, const float* __restrict__ Bw, CT* __restrict__ C,
    int M, int N, int K)
{
  __shared__ __align__(16) _Float16 As[2][128][40]; // [buf][m][k], padded stride
  __shared__ __align__(16) _Float16 Bs[2][128][40]; // [buf][n][k] (transposed)

  const int tid  = threadIdx.x;
  const int lane = tid & 31;
  const int l15  = lane & 15;
  const int lhi  = lane >> 4;
  const int wave = __builtin_amdgcn_readfirstlane(tid >> 5);  // scalar
  const int wm   = (wave & 3) * 32;   // wave M offset in block tile
  const int wn   = (wave >> 2) * 64;  // wave N offset in block tile
  const int bm   = blockIdx.y * 128;
  const int bn   = blockIdx.x * 128;

  const int ar  = tid >> 3;        // A tile row base (0..31), +32*p
  const int ak  = (tid & 7) * 4;   // A tile k col
  const int bk  = tid >> 5;        // B tile k row base (0..7), +8*p
  const int bn4 = (tid & 31) * 4;  // B tile n col

  v4h ta[4], tb[4];  // staged (already f16-converted) tile data

  auto fetch = [&](int k0) {
#pragma unroll
    for (int p = 0; p < 4; ++p) {
      const AT* srcA = A + (size_t)(bm + ar + p * 32) * K + (k0 + ak);
      if constexpr (__is_same(AT, float)) {
        v4f t = *(const v4f*)srcA;
#pragma unroll
        for (int j = 0; j < 4; ++j) ta[p][j] = (_Float16)t[j];
      } else {
        ta[p] = *(const v4h*)srcA;
      }
      v4f t2 = *(const v4f*)(Bw + (size_t)(k0 + bk + p * 8) * N + (bn + bn4));
#pragma unroll
      for (int j = 0; j < 4; ++j) tb[p][j] = (_Float16)t2[j];
    }
  };
  auto stage = [&](int buf) {
#pragma unroll
    for (int p = 0; p < 4; ++p) {
      *(v4h*)&As[buf][ar + p * 32][ak] = ta[p];
#pragma unroll
      for (int j = 0; j < 4; ++j) Bs[buf][bn4 + j][bk + p * 8] = tb[p][j];
    }
  };

  v8f acc[2][4] = {};
  fetch(0);
  stage(0);
  int cur = 0;
  for (int k0 = 0; k0 < K; k0 += 32) {
    __syncthreads();                       // buf[cur] ready for everyone
    const bool more = (k0 + 32) < K;       // scalar
    if (more) fetch(k0 + 32);              // global loads overlap compute below

    // ---- fragments (match ISA 16-bit A 16x32 / B 32x16 VGPR layouts) ----
    v16h afrag[2], bfrag[4];
#pragma unroll
    for (int mi = 0; mi < 2; ++mi) {
      const _Float16* ap = &As[cur][wm + mi * 16 + l15][lhi * 8];
      v8h lo = *(const v8h*)ap;         // K = lhi*8 .. +7
      v8h hi = *(const v8h*)(ap + 16);  // K = 16 + lhi*8 .. +7
      afrag[mi] = combine16(lo, hi);
    }
#pragma unroll
    for (int ni = 0; ni < 4; ++ni) {
      const _Float16* bp = &Bs[cur][wn + ni * 16 + l15][lhi * 8];
      v8h lo = *(const v8h*)bp;         // K = lhi*8 .. +7
      v8h hi = *(const v8h*)(bp + 16);  // K = 16 + lhi*8 .. +7
      bfrag[ni] = combine16(lo, hi);
    }
#pragma unroll
    for (int mi = 0; mi < 2; ++mi)
#pragma unroll
      for (int ni = 0; ni < 4; ++ni)
        acc[mi][ni] = wmma_f16(afrag[mi], bfrag[ni], acc[mi][ni]);

    if (more) stage(cur ^ 1);              // fill other buffer
    cur ^= 1;
  }

  // ---- epilogue: C layout = lane holds col l15, VGPR i = row i + 8*lhi ----
#pragma unroll
  for (int mi = 0; mi < 2; ++mi)
#pragma unroll
    for (int ni = 0; ni < 4; ++ni)
#pragma unroll
      for (int i = 0; i < 8; ++i) {
        int row = bm + wm + mi * 16 + i + 8 * lhi;
        int col = bn + wn + ni * 16 + l15;
        C[(size_t)row * N + col] = (CT)acc[mi][ni][i];
      }
}

// ---------------------------------------------------------------------------
// RoPE + head split:  qkv f16 [B*S, H*384] ->
//   Qh [B,H,S,HD] (rope, * 1/sqrt(HD)), Kh [B,H,S,HD] (rope), Vt [B,H,HD,S]
// ---------------------------------------------------------------------------
__global__ __launch_bounds__(256) void rope_split_kernel(
    const _Float16* __restrict__ qkv, const float* __restrict__ rsin,
    const float* __restrict__ rcos, _Float16* __restrict__ Qh,
    _Float16* __restrict__ Kh, _Float16* __restrict__ Vt)
{
  int idx = blockIdx.x * 256 + threadIdx.x;  // over B*S*H*HD = 2^23
  int d = idx & 127;
  int h = (idx >> 7) & 15;
  int s = (idx >> 11) & 2047;
  int b = idx >> 22;
  const _Float16* base = qkv + (size_t)(b * S_ + s) * (H_ * 3 * HD_) + h * (3 * HD_);
  size_t bh = (size_t)(b * H_ + h);
  // V: transpose into [B,H,HD,S]
  Vt[(bh * HD_ + d) * S_ + s] = base[2 * HD_ + d];
  if (d < 64) {
    float sn = rsin[s * 64 + d];
    float cs = rcos[s * 64 + d];
    float q1 = (float)base[d],        q2 = (float)base[64 + d];
    float k1 = (float)base[HD_ + d],  k2 = (float)base[HD_ + 64 + d];
    const float qs = 0.08838834764831845f;  // 1/sqrt(128)
    size_t o = (bh * S_ + s) * HD_;
    Qh[o + d]      = (_Float16)((q1 * cs - q2 * sn) * qs);
    Qh[o + 64 + d] = (_Float16)((q2 * cs + q1 * sn) * qs);
    Kh[o + d]      = (_Float16)(k1 * cs - k2 * sn);
    Kh[o + 64 + d] = (_Float16)(k2 * cs + k1 * sn);
  }
}

// ---------------------------------------------------------------------------
// Flash attention with softcap + causal mask.
// One wave per 16-row q tile; 4 waves/block. Q frags in registers.
// All wave-uniform control values go through readfirstlane -> scalar branches
// (EXEC untouched around WMMA). Row sums computed with a WMMA against an
// all-ones B matrix instead of cross-lane shuffle reductions.
// Output X: [B,S,H*HD] f16.
// ---------------------------------------------------------------------------
__global__ __launch_bounds__(128) void attn_kernel(
    const _Float16* __restrict__ Q, const _Float16* __restrict__ K,
    const _Float16* __restrict__ V, _Float16* __restrict__ X)
{
  __shared__ __align__(16) _Float16 Ps[4][16][32];
  const int tid  = threadIdx.x;
  const int lane = tid & 31;
  const int l15  = lane & 15;
  const int lhi  = lane >> 4;
  const int wave = __builtin_amdgcn_readfirstlane(tid >> 5);  // scalar
  const int w    = blockIdx.x * 4 + wave;                     // scalar
  const int qt   = w & (S_ / 16 - 1);
  const int bh   = w >> 7;  // b*H + h
  const int q0   = qt * 16;
  const size_t bhS = (size_t)bh * S_;

  v16h ones;
#pragma unroll
  for (int i = 0; i < 16; ++i) ones[i] = (_Float16)1.0f;

  // Q A-fragments for full HD=128 (4 chunks of K=32)
  v16h qf[4];
  {
    const _Float16* qb = Q + (bhS + q0 + l15) * HD_;
#pragma unroll
    for (int kc = 0; kc < 4; ++kc) {
      v8h lo = *(const v8h*)(qb + kc * 32 + lhi * 8);
      v8h hi = *(const v8h*)(qb + kc * 32 + 16 + lhi * 8);
      qf[kc] = combine16(lo, hi);
    }
  }

  v8f o[8] = {};
  float mrow[8], lrow[8];
#pragma unroll
  for (int i = 0; i < 8; ++i) { mrow[i] = -1e30f; lrow[i] = 0.f; }

  const int kmax = q0 + 16;             // #valid key columns (causal), scalar
  const int nch  = (kmax + 31) >> 5;    // 32-wide chunks, scalar
  for (int kb = 0; kb < nch; ++kb) {
    v8f sv[2];
    float chm[8];
#pragma unroll
    for (int i = 0; i < 8; ++i) chm[i] = -1e30f;

#pragma unroll
    for (int kt = 0; kt < 2; ++kt) {
      int kc0 = kb * 32 + kt * 16;      // scalar
      v8f s = {};
      if (kc0 < kmax) {                 // scalar branch, EXEC untouched
        const _Float16* kbp = K + (bhS + kc0 + l15) * HD_;
        v16h kf[4];                     // hoist all loads before WMMA chain
#pragma unroll
        for (int kc = 0; kc < 4; ++kc) {
          v8h lo = *(const v8h*)(kbp + kc * 32 + lhi * 8);
          v8h hi = *(const v8h*)(kbp + kc * 32 + 16 + lhi * 8);
          kf[kc] = combine16(lo, hi);
        }
#pragma unroll
        for (int kc = 0; kc < 4; ++kc) s = wmma_f16(qf[kc], kf[kc], s);
        int col = kc0 + l15;
#pragma unroll
        for (int i = 0; i < 8; ++i) {
          float sc = 50.f * tanhf(s[i] * 0.02f);     // softcap
          int row = q0 + i + 8 * lhi;
          s[i] = (col <= row) ? sc : -1e30f;          // causal mask
        }
#pragma unroll
        for (int i = 0; i < 8; ++i) {                 // row max over 16 cols
          float v = s[i];
#pragma unroll
          for (int off = 1; off < 16; off <<= 1) v = fmaxf(v, __shfl_xor(v, off, 16));
          chm[i] = fmaxf(chm[i], v);
        }
      } else {
#pragma unroll
        for (int i = 0; i < 8; ++i) s[i] = -1e30f;    // chm stays -1e30
      }
      sv[kt] = s;
    }

    float osc[8];
#pragma unroll
    for (int i = 0; i < 8; ++i) {
      float nm = fmaxf(mrow[i], chm[i]);
      osc[i] = __expf(mrow[i] - nm);
      mrow[i] = nm;
      lrow[i] *= osc[i];
    }
#pragma unroll
    for (int nt = 0; nt < 8; ++nt)
#pragma unroll
      for (int i = 0; i < 8; ++i) o[nt][i] *= osc[i];

#pragma unroll
    for (int kt = 0; kt < 2; ++kt)
#pragma unroll
      for (int i = 0; i < 8; ++i) {
        float p = __expf(sv[kt][i] - mrow[i]);        // masked -> exp(-1e30)=0
        Ps[wave][i + 8 * lhi][kt * 16 + l15] = (_Float16)p;
      }

    // P: C-layout -> A-layout via LDS (same-wave DS ops are in-order)
    v8h plo = *(const v8h*)&Ps[wave][l15][lhi * 8];
    v8h phi = *(const v8h*)&Ps[wave][l15][16 + lhi * 8];
    v16h pa = combine16(plo, phi);

    // row sums of P via WMMA with all-ones B (replaces shuffle reduction)
    {
      v8f zero = {};
      v8f csum = wmma_f16(pa, ones, zero);
#pragma unroll
      for (int i = 0; i < 8; ++i) lrow[i] += csum[i];
    }

#pragma unroll
    for (int nt = 0; nt < 8; ++nt) {
      const _Float16* vp = V + ((size_t)bh * HD_ + nt * 16 + l15) * S_ + kb * 32 + lhi * 8;
      v8h vlo = *(const v8h*)vp;
      v8h vhi = *(const v8h*)(vp + 16);
      o[nt] = wmma_f16(pa, combine16(vlo, vhi), o[nt]);
    }
  }

  float inv[8];
#pragma unroll
  for (int i = 0; i < 8; ++i) inv[i] = 1.f / lrow[i];
  const int b = bh >> 4, h = bh & 15;
#pragma unroll
  for (int nt = 0; nt < 8; ++nt)
#pragma unroll
    for (int i = 0; i < 8; ++i) {
      int row = q0 + i + 8 * lhi;
      int col = h * HD_ + nt * 16 + l15;
      X[(size_t)(b * S_ + row) * D_ + col] = (_Float16)(o[nt][i] * inv[i]);
    }
}

// ---------------------------------------------------------------------------
extern "C" void kernel_launch(void* const* d_in, const int* in_sizes, int n_in,
                              void* d_out, int out_size, void* d_ws, size_t ws_size,
                              hipStream_t stream) {
  (void)in_sizes; (void)n_in; (void)out_size; (void)ws_size;
  const float* inputs   = (const float*)d_in[0];  // [B,S,D]
  const float* w_in     = (const float*)d_in[1];  // [D,H,3*HD] == [D, 6144]
  const float* w_out    = (const float*)d_in[2];  // [H,HD,D]  == [2048, D]
  const float* rope_sin = (const float*)d_in[3];  // [S,64]
  const float* rope_cos = (const float*)d_in[4];  // [S,64]
  // d_in[5] = mask: causal, computed analytically in-kernel
  float* out = (float*)d_out;                     // [B,S,D] f32

  const size_t QKV_ELEMS  = (size_t)B_ * S_ * H_ * 3 * HD_;  // 25,165,824
  const size_t HEAD_ELEMS = (size_t)B_ * H_ * S_ * HD_;      //  8,388,608
  _Float16* qkv = (_Float16*)d_ws;
  _Float16* Qh  = qkv + QKV_ELEMS;
  _Float16* Kh  = Qh + HEAD_ELEMS;
  _Float16* Vt  = Kh + HEAD_ELEMS;
  _Float16* X   = Vt + HEAD_ELEMS;

  // 1) QKV projection: (B*S=4096, K=2048) x (2048, 6144) -> qkv f16
  gemm_wmma_kernel<float, _Float16><<<dim3(6144 / 128, 4096 / 128), 256, 0, stream>>>(
      inputs, w_in, qkv, 4096, 6144, 2048);
  // 2) RoPE + split + V transpose
  rope_split_kernel<<<(B_ * S_ * H_ * HD_) / 256, 256, 0, stream>>>(
      qkv, rope_sin, rope_cos, Qh, Kh, Vt);
  // 3) Flash attention (4096 q-tiles, 4 waves/block)
  attn_kernel<<<(B_ * H_ * (S_ / 16)) / 4, 128, 0, stream>>>(Qh, Kh, Vt, X);
  // 4) Output projection: (4096, 2048) x (2048, 2048) -> out f32
  gemm_wmma_kernel<_Float16, float><<<dim3(2048 / 128, 4096 / 128), 256, 0, stream>>>(
      X, w_out, out, 4096, 2048, 2048);
}
